// SelfAttention_31619549234003
// MI455X (gfx1250) — matrix operand order repaired and checked by reading, not compiled, not run on hardware
//
#include <hip/hip_runtime.h>
#include <hip/hip_bf16.h>

// ---------------------------------------------------------------------------
// Self-attention (B=1, S=4096, H=1024, NH=16, HD=64), fp32 in/out.
// QKV GEMMs + flash attention on v_wmma_f32_16x16x32_f16 (f32 accumulate).
// GEMM B-operand staged via async global->LDS DMA (ASYNCcnt), double-buffered.
// ---------------------------------------------------------------------------

#define S_LEN 4096
#define HDIM  1024
#define NHEAD 16
#define HD    64
// softmax scale (HD^-0.5 = 0.125) * log2(e), folded into Q so we can use exp2
#define QSCALE 0.18033688011112042f
#define L2_10000 13.287712379549449f   // log2(10000)

typedef _Float16 v8h  __attribute__((ext_vector_type(8)));
typedef _Float16 v16h __attribute__((ext_vector_type(16)));
typedef float    v8f  __attribute__((ext_vector_type(8)));

__device__ __forceinline__ v16h ld_v16h(const _Float16* p) {
    return *(const v16h*)p;
}
// assemble a 16-half fragment from two contiguous 8-half chunks
__device__ __forceinline__ v16h ld2_v8h(const _Float16* plo, const _Float16* phi) {
    v8h lo = *(const v8h*)plo;
    v8h hi = *(const v8h*)phi;
    v16h r;
#pragma unroll
    for (int i = 0; i < 8; ++i) { r[i] = lo[i]; r[i + 8] = hi[i]; }
    return r;
}

// async global -> LDS 16-byte copy (no VGPR transit; tracked by ASYNCcnt)
__device__ __forceinline__ void async_copy_b128(const void* gsrc, void* lds_dst) {
    const unsigned long long ga = (unsigned long long)(uintptr_t)gsrc;
    const unsigned int       la = (unsigned int)(uintptr_t)lds_dst; // low 32b = LDS offset
    asm volatile("global_load_async_to_lds_b128 %0, %1, off"
                 :: "v"(la), "v"(ga) : "memory");
}
__device__ __forceinline__ void wait_asynccnt_le1() {
    asm volatile("s_wait_asynccnt 0x1" ::: "memory");
}
__device__ __forceinline__ void wait_asynccnt_0() {
    asm volatile("s_wait_asynccnt 0x0" ::: "memory");
}

// ---------------------------------------------------------------------------
// Kernel 1: fp32 -> fp16 conversion (hidden states + weights)
// ---------------------------------------------------------------------------
__global__ void __launch_bounds__(256)
cvt_f32_f16(const float* __restrict__ src, _Float16* __restrict__ dst, int n) {
    for (int i = blockIdx.x * blockDim.x + threadIdx.x; i < n;
         i += gridDim.x * blockDim.x)
        dst[i] = (_Float16)src[i];
}

// ---------------------------------------------------------------------------
// Kernel 2: Y[s, o] = sum_k X[s, k] * W[o, k] + bias[o]   (x @ W.T + b)
// M=4096, N=1024, K=1024.  Block = 256 threads = 8 waves.
// Wave tile: 16 (M) x 64 (N) -> 4 accumulators, A-fragment reused 4x.
// Block tile: 128 x 64.  W tile (64 x 32 f16 = 4 KB) staged in LDS by
// async DMA, double-buffered: next tile's copy overlaps current WMMAs.
// ---------------------------------------------------------------------------
__global__ void __launch_bounds__(256)
qkv_gemm(const _Float16* __restrict__ X, const _Float16* __restrict__ W,
         const float* __restrict__ bias, float* __restrict__ Y) {
    __shared__ _Float16 wtile[2][64][32];   // [buf][o_local][k_local], 8 KB

    const int tid  = threadIdx.x;
    const int lane = tid & 31;
    const int wave = tid >> 5;
    const int half = lane >> 4;     // which 16-lane half of the wave32
    const int lm   = lane & 15;

    const int s0 = blockIdx.y * 128 + wave * 16;   // query-row base
    const int o0 = blockIdx.x * 64;                // output-col base

    // this thread's 16-byte slice of the 4 KB W tile
    const int co = tid >> 2;          // 0..63  (o_local)
    const int ck = (tid & 3) * 8;     // 0,8,16,24 (k_local)
    const _Float16* wsrc = W + (size_t)(o0 + co) * HDIM + ck;

    v8f acc[4] = {};
    const _Float16* xrow = X + (size_t)(s0 + lm) * HDIM;

    // prologue: DMA first W tile into buffer 0
    async_copy_b128(wsrc, &wtile[0][co][ck]);

    for (int k = 0; k < HDIM; k += 32) {
        const int buf = (k >> 5) & 1;
        const bool more = (k + 32) < HDIM;
        if (more) {
            // overlap: DMA next tile into the other buffer
            async_copy_b128(wsrc + k + 32, &wtile[buf ^ 1][co][ck]);
            wait_asynccnt_le1();      // current tile's DMA (older) is done
        } else {
            wait_asynccnt_0();
        }
        __syncthreads();              // all waves' slices landed in LDS

        // A fragment (16x32 f16): lane = row, 16 contiguous K at 16*half
        v16h a = ld2_v8h(xrow + k + 16 * half, xrow + k + 16 * half + 8);
#pragma unroll
        for (int t = 0; t < 4; ++t) {
            // B fragment (32x16 f16): lane lm = column = o-local 16t+lm,
            // 16 contiguous K values read from LDS
            v16h b = *(const v16h*)&wtile[buf][16 * t + lm][16 * half];
            acc[t] = __builtin_amdgcn_wmma_f32_16x16x32_f16(
                false, a, false, b, (short)0, acc[t], false, false);
        }
        __syncthreads();              // protect buf^1 before next overwrite
    }

    // Epilogue: C layout -> row = r + 8*half, col = o0 + 16t + lm
#pragma unroll
    for (int t = 0; t < 4; ++t) {
        const int o = o0 + 16 * t + lm;
        const float bv = bias[o];
#pragma unroll
        for (int r = 0; r < 8; ++r) {
            const int srow = s0 + r + 8 * half;
            Y[(size_t)srow * HDIM + o] = acc[t][r] + bv;
        }
    }
}

// ---------------------------------------------------------------------------
// Kernel 3: RoPE on Q,K + repack to f16.
//   Qh [NH][S][HD]  (scaled by QSCALE),  Kh [NH][S][HD],  Vt [NH][HD][S]
// ---------------------------------------------------------------------------
__global__ void __launch_bounds__(256)
rope_pack(const float* __restrict__ Yq, const float* __restrict__ Yk,
          const float* __restrict__ Yv, _Float16* __restrict__ Qh,
          _Float16* __restrict__ Kh, _Float16* __restrict__ Vt) {
    const int n = NHEAD * S_LEN * HD;
    for (int idx = blockIdx.x * blockDim.x + threadIdx.x; idx < n;
         idx += gridDim.x * blockDim.x) {
        const int d = idx & (HD - 1);
        const int s = (idx >> 6) & (S_LEN - 1);
        const int h = idx >> 18;
        const int o = h * HD + d;
        const size_t yi = (size_t)s * HDIM + o;
        const size_t yp = (size_t)s * HDIM + (o ^ 32);   // rotate-half partner

        const int   fi  = d & 31;
        const float inv = exp2f(-(float)fi * (L2_10000 / 32.0f)); // 10000^(-fi/32)
        const float ang = (float)s * inv;
        float sn, cs;
        __sincosf(ang, &sn, &cs);

        const float q = Yq[yi], qp = Yq[yp];
        const float k = Yk[yi], kp = Yk[yp];
        const float rsign = (d < 32) ? -1.0f : 1.0f;
        const float qr = q * cs + rsign * qp * sn;
        const float kr = k * cs + rsign * kp * sn;

        Qh[((size_t)h * S_LEN + s) * HD + d] = (_Float16)(qr * QSCALE);
        Kh[((size_t)h * S_LEN + s) * HD + d] = (_Float16)kr;
        Vt[((size_t)h * HD + d) * S_LEN + s] = (_Float16)Yv[yi];
    }
}

// ---------------------------------------------------------------------------
// Kernel 4: flash attention.  Block = 128 threads = 4 waves = 1 head x 64 q.
// Each wave: 16 query rows, streams keys in chunks of 32 with online softmax.
// Scores in log2 domain (scale folded into Q) -> native v_exp_f32.
// ---------------------------------------------------------------------------
__global__ void __launch_bounds__(128)
flash_attn(const _Float16* __restrict__ Qh, const _Float16* __restrict__ Kh,
           const _Float16* __restrict__ Vt, const float* __restrict__ mask,
           float* __restrict__ out) {
    __shared__ _Float16 pbuf[4][16][32];   // wave-private P staging tiles

    const int lane = threadIdx.x & 31;
    const int wave = threadIdx.x >> 5;
    const int half = lane >> 4;
    const int lm   = lane & 15;

    const int h  = blockIdx.y;
    const int q0 = blockIdx.x * 64 + wave * 16;

    const _Float16* Qrow  = Qh + ((size_t)h * S_LEN + q0) * HD + lm * HD;
    const _Float16* Kbase = Kh + (size_t)h * S_LEN * HD;
    const _Float16* Vbase = Vt + (size_t)h * HD * S_LEN;

    // Q A-fragments for hd chunks [0,32) and [32,64) — kept in registers
    v16h qa[2];
#pragma unroll
    for (int j = 0; j < 2; ++j)
        qa[j] = ld2_v8h(Qrow + 32 * j + 16 * half, Qrow + 32 * j + 16 * half + 8);

    float mrun[8], lsum[8];
    v8f acc[4] = {};
#pragma unroll
    for (int r = 0; r < 8; ++r) { mrun[r] = -3.0e38f; lsum[r] = 0.0f; }

    for (int key0 = 0; key0 < S_LEN; key0 += 32) {
        // prefetch next key/value chunk into L0 (global_prefetch_b8)
        if (key0 + 32 < S_LEN) {
            __builtin_prefetch(Kbase + (size_t)(key0 + 32 + lm) * HD, 0, 0);
            __builtin_prefetch(Vbase + (size_t)(16 + lm) * S_LEN + key0 + 32, 0, 0);
        }

        // ---- scores: two 16x16 tiles, contraction over HD=64 (2 WMMAs each)
        v8f sc0 = {}, sc1 = {};
#pragma unroll
        for (int j = 0; j < 2; ++j) {
            v16h b0 = ld_v16h(Kbase + (size_t)(key0 + lm) * HD
                              + 32 * j + 16 * half);
            v16h b1 = ld_v16h(Kbase + (size_t)(key0 + 16 + lm) * HD
                              + 32 * j + 16 * half);
            sc0 = __builtin_amdgcn_wmma_f32_16x16x32_f16(
                false, qa[j], false, b0, (short)0, sc0, false, false);
            sc1 = __builtin_amdgcn_wmma_f32_16x16x32_f16(
                false, qa[j], false, b1, (short)0, sc1, false, false);
        }
        // additive mask (broadcast over queries); convert to log2 domain
        const float mk0 = mask[key0 + lm] * 1.44269504f;
        const float mk1 = mask[key0 + 16 + lm] * 1.44269504f;

        // ---- online softmax (rows live per-vgpr; reduce across 16-lane half)
        float p0[8], p1[8];
#pragma unroll
        for (int r = 0; r < 8; ++r) {
            const float a = sc0[r] + mk0;
            const float b = sc1[r] + mk1;
            float cm = fmaxf(a, b);
#pragma unroll
            for (int m = 1; m < 16; m <<= 1)
                cm = fmaxf(cm, __shfl_xor(cm, m, 32));
            const float mnew = fmaxf(mrun[r], cm);
            const float corr = exp2f(mrun[r] - mnew);
            const float e0 = exp2f(a - mnew);
            const float e1 = exp2f(b - mnew);
            float rs = e0 + e1;
#pragma unroll
            for (int m = 1; m < 16; m <<= 1)
                rs += __shfl_xor(rs, m, 32);
            lsum[r] = lsum[r] * corr + rs;
            mrun[r] = mnew;
#pragma unroll
            for (int t = 0; t < 4; ++t) acc[t][r] *= corr;
            p0[r] = e0;
            p1[r] = e1;
        }

        // ---- P: C layout -> A layout via wave-private LDS tile
#pragma unroll
        for (int r = 0; r < 8; ++r) {
            const int m = r + 8 * half;
            pbuf[wave][m][lm]      = (_Float16)p0[r];
            pbuf[wave][m][lm + 16] = (_Float16)p1[r];
        }
        // DS ops are in-order within a wave; loads below wait on DScnt.
        const _Float16* pp = &pbuf[wave][lm][16 * half];
        v16h pa = ld2_v8h(pp, pp + 8);

        // ---- acc += P (16x32) x V (32x16 per hd tile)
#pragma unroll
        for (int t = 0; t < 4; ++t) {
            v16h vb = ld_v16h(Vbase + (size_t)(16 * t + lm) * S_LEN
                              + key0 + 16 * half);
            acc[t] = __builtin_amdgcn_wmma_f32_16x16x32_f16(
                false, pa, false, vb, (short)0, acc[t], false, false);
        }
    }

    // ---- normalize + store: out[B,S,H] fp32
#pragma unroll
    for (int r = 0; r < 8; ++r) {
        const float invl = 1.0f / lsum[r];
        const int srow = q0 + r + 8 * half;
#pragma unroll
        for (int t = 0; t < 4; ++t)
            out[(size_t)srow * HDIM + h * HD + 16 * t + lm] = acc[t][r] * invl;
    }
}

// ---------------------------------------------------------------------------
// Host launcher.  Workspace layout (bytes):
//   Xh   f16 [S][H]        0        ..  8 MB
//   Wqh/Wkh/Wvh f16 [H][H] 8 MB     .. 14 MB
//   Yq/Yk/Yv   f32 [S][H]  14 MB    .. 62 MB
//   Qh/Kh f16 [NH][S][HD], Vt f16 [NH][HD][S]: 62 MB .. 86 MB
// ---------------------------------------------------------------------------
extern "C" void kernel_launch(void* const* d_in, const int* in_sizes, int n_in,
                              void* d_out, int out_size, void* d_ws, size_t ws_size,
                              hipStream_t stream) {
    const float* X    = (const float*)d_in[0];
    const float* mask = (const float*)d_in[1];
    const float* Wq   = (const float*)d_in[2];
    const float* bq   = (const float*)d_in[3];
    const float* Wk   = (const float*)d_in[4];
    const float* bk   = (const float*)d_in[5];
    const float* Wv   = (const float*)d_in[6];
    const float* bv   = (const float*)d_in[7];
    float* out = (float*)d_out;

    char* ws = (char*)d_ws;
    const size_t SZ_XH = (size_t)S_LEN * HDIM * 2;   // 8 MB
    const size_t SZ_W  = (size_t)HDIM * HDIM * 2;    // 2 MB
    const size_t SZ_Y  = (size_t)S_LEN * HDIM * 4;   // 16 MB
    const size_t SZ_QH = (size_t)NHEAD * S_LEN * HD * 2; // 8 MB

    _Float16* Xh  = (_Float16*)(ws);
    _Float16* Wqh = (_Float16*)(ws + SZ_XH);
    _Float16* Wkh = (_Float16*)(ws + SZ_XH + SZ_W);
    _Float16* Wvh = (_Float16*)(ws + SZ_XH + 2 * SZ_W);
    float*    Yq  = (float*)(ws + SZ_XH + 3 * SZ_W);
    float*    Yk  = (float*)(ws + SZ_XH + 3 * SZ_W + SZ_Y);
    float*    Yv  = (float*)(ws + SZ_XH + 3 * SZ_W + 2 * SZ_Y);
    _Float16* Qh  = (_Float16*)(ws + SZ_XH + 3 * SZ_W + 3 * SZ_Y);
    _Float16* Kh  = (_Float16*)(ws + SZ_XH + 3 * SZ_W + 3 * SZ_Y + SZ_QH);
    _Float16* Vt  = (_Float16*)(ws + SZ_XH + 3 * SZ_W + 3 * SZ_Y + 2 * SZ_QH);

    // 1) f32 -> f16 operand conversion
    cvt_f32_f16<<<2048, 256, 0, stream>>>(X, Xh, S_LEN * HDIM);
    cvt_f32_f16<<<1024, 256, 0, stream>>>(Wq, Wqh, HDIM * HDIM);
    cvt_f32_f16<<<1024, 256, 0, stream>>>(Wk, Wkh, HDIM * HDIM);
    cvt_f32_f16<<<1024, 256, 0, stream>>>(Wv, Wvh, HDIM * HDIM);

    // 2) QKV projections (WMMA GEMM, async-LDS staged B)
    dim3 ggrid(HDIM / 64, S_LEN / 128);
    qkv_gemm<<<ggrid, 256, 0, stream>>>(Xh, Wqh, bq, Yq);
    qkv_gemm<<<ggrid, 256, 0, stream>>>(Xh, Wkh, bk, Yk);
    qkv_gemm<<<ggrid, 256, 0, stream>>>(Xh, Wvh, bv, Yv);

    // 3) RoPE + f16 repack (scale folded into Q)
    rope_pack<<<4096, 256, 0, stream>>>(Yq, Yk, Yv, Qh, Kh, Vt);

    // 4) flash attention + output
    dim3 agrid(S_LEN / 64, NHEAD);
    flash_attn<<<agrid, 128, 0, stream>>>(Qh, Kh, Vt, mask, out);
}
